// CosformerAttention_82841329205679
// MI455X (gfx1250) — matrix operand order repaired and checked by reading, not compiled, not run on hardware
//
#include <hip/hip_runtime.h>

// CDNA5 / gfx1250 cosformer causal linear attention, chunked WMMA formulation.
// B=8, T=4096, D=128, phi-features F=256, chunk C=32, dv tile = 16.

typedef _Float16 h16;
typedef __attribute__((ext_vector_type(16))) _Float16 v16h;
typedef __attribute__((ext_vector_type(8)))  float    v8f;

#define WMMA_F32_F16(a, b, c) \
  __builtin_amdgcn_wmma_f32_16x16x32_f16(false, (a), false, (b), (short)0, (c), false, false)

#define B_   8
#define T_   4096
#define D_   128
#define F_   256        // phi feature dim = 2*D ... wait: 2*128 = 256
#define C_   32         // time chunk
#define NT_  (T_ / C_)
#define DVT_ 16         // dv tile width per workgroup

union Frag16 {
  v16h v;
  h16 e[16];
  unsigned u[8];
};

// A-operand (16x32 f16), source staged row-major with given stride.
// Layout: lanes 0-15 -> M=lane, K=0..15 ; lanes 16-31 -> K=16..31 (contiguous, matches B).
__device__ inline v16h load_a_rowmajor(const h16* lds, int row0, int col0, int stride, int lane) {
  int m = lane & 15;
  int kh = (lane >> 4) * 16;
  const h16* base = lds + (row0 + m) * stride + col0;
  Frag16 f;
#pragma unroll
  for (int vv = 0; vv < 8; ++vv) {
    int kk = 2 * vv + kh;
    f.u[vv] = *(const unsigned*)(base + kk);   // aligned pair (K, K+1)
  }
  return f.v;
}

// B-operand (32x16 f16), source staged column-major: element B[k][n] at lds[n*stride + k].
// Layout: lanes 0-15 -> N=lane, K=0..15 over v0..7 (pairs); lanes 16-31 -> K=16..31.
__device__ inline v16h load_b_colmajor(const h16* lds, int n0, int k0, int stride, int lane) {
  int n = lane & 15;
  int koff = (lane >> 4) * 16;
  const h16* base = lds + (n0 + n) * stride + k0 + koff;
  Frag16 f;
#pragma unroll
  for (int vv = 0; vv < 8; ++vv) f.u[vv] = *(const unsigned*)(base + 2 * vv);
  return f.v;
}

// B-operand for Kf^T where the staging is kft[f*32 + j] (row-major in f): strided gather.
__device__ inline v16h load_b_kft(const h16* kft, int n0, int k0, int lane) {
  int n = n0 + (lane & 15);
  int koff = (lane >> 4) * 16;
  Frag16 f;
#pragma unroll
  for (int vv = 0; vv < 8; ++vv) {
    int fi = k0 + koff + 2 * vv;
    f.e[2 * vv]     = kft[fi * C_ + n];
    f.e[2 * vv + 1] = kft[(fi + 1) * C_ + n];
  }
  return f.v;
}

// A-operand tile of (kf^T scaled by per-time weight w_k), built from kft[f*32+k] staging.
__device__ inline v16h load_a_kft_scaled(const h16* kft, int fr, const float* w, int lane) {
  int m = lane & 15;
  int kh = (lane >> 4) * 16;
  const h16* base = kft + (fr + m) * C_;
  Frag16 f;
#pragma unroll
  for (int vv = 0; vv < 8; ++vv) {
    int kk = 2 * vv + kh;
    f.e[2 * vv]     = base[kk]     * (h16)w[kk];
    f.e[2 * vv + 1] = base[kk + 1] * (h16)w[kk + 1];
  }
  return f.v;
}

__global__ __launch_bounds__(256)
void cosformer_chunked_wmma(const float* __restrict__ q,
                            const float* __restrict__ k,
                            const float* __restrict__ v,
                            float* __restrict__ out) {
  __shared__ h16 sQf[C_ * F_];       // qf row-major (i*F + f)            16 KB
  __shared__ h16 sKft[F_ * C_];      // kf^T row-major (f*C + i)          16 KB
  __shared__ h16 sVt[DVT_ * C_];     // V^T col-major staging (n*C + i)    1 KB
  __shared__ h16 sSc[DVT_ * F_];     // f16 shadow of S_cos (n*F + f)      8 KB
  __shared__ h16 sSs[DVT_ * F_];     // f16 shadow of S_sin                8 KB
  __shared__ h16 sA[C_ * C_];        // masked weighted intra scores       2 KB
  __shared__ float sCos[C_], sSin[C_];
  __shared__ float sZc[F_], sZs[F_];
  __shared__ float sDen[C_];

  const int blk  = blockIdx.x;
  const int b    = blk >> 3;   // batch
  const int dt   = blk & 7;    // dv tile
  const int tid  = threadIdx.x;
  const int wave = tid >> 5;
  const int lane = tid & 31;

  // VGPR-resident f32 state accumulators: wave owns feature rows [wave*32, wave*32+32)
  v8f Sc[2], Ss[2];
#pragma unroll
  for (int i = 0; i < 2; ++i) { Sc[i] = {}; Ss[i] = {}; }

  if (tid < F_) { sZc[tid] = 0.f; sZs[tid] = 0.f; }
  for (int i = tid; i < DVT_ * F_; i += 256) { sSc[i] = (h16)0.f; sSs[i] = (h16)0.f; }

  const float thscale = 3.14159265358979323846f / (2.0f * (float)T_);

  for (int ct = 0; ct < NT_; ++ct) {
    const int t0 = ct * C_;
    __syncthreads();  // protect phase-0 writes vs previous iteration's phase-3 reads

    // ---------------- Phase 0: stage chunk ----------------
    {
      int i  = tid >> 3;            // chunk row 0..31 (8 threads per row)
      int c0 = (tid & 7) * 16;      // feature column start
      if (tid < C_) {
        float tht = thscale * (float)(t0 + tid);
        sCos[tid] = __cosf(tht);
        sSin[tid] = __sinf(tht);
        sDen[tid] = 0.f;
      }
      const size_t rowoff = ((size_t)b * T_ + (size_t)(t0 + i)) * D_;
      const float* qr = q + rowoff + c0;
      const float* kr = k + rowoff + c0;
#pragma unroll
      for (int u = 0; u < 16; ++u) {
        float qv = qr[u];
        sQf[i * F_ + c0 + u]      = (h16)fmaxf(qv, 0.f);
        sQf[i * F_ + D_ + c0 + u] = (h16)fmaxf(-qv, 0.f);
        float kv = kr[u];
        sKft[(c0 + u) * C_ + i]      = (h16)fmaxf(kv, 0.f);
        sKft[(D_ + c0 + u) * C_ + i] = (h16)fmaxf(-kv, 0.f);
      }
      int cv0 = (tid & 7) * 2;      // 2 dv columns per thread
      const float* vr = v + rowoff + dt * DVT_ + cv0;
#pragma unroll
      for (int u = 0; u < 2; ++u) sVt[(cv0 + u) * C_ + i] = (h16)vr[u];
    }
    __syncthreads();

    // ---- Phase 1: intra scores A = Qf Kf^T (waves 0-3); inter denominator (waves 4-7) ----
    if (wave < 4) {
      int mi = (wave >> 1) * 16, ni = (wave & 1) * 16;
      v8f acc = {};
#pragma unroll
      for (int kt = 0; kt < 8; ++kt) {
        v16h a  = load_a_rowmajor(sQf, mi, kt * 32, F_, lane);
        v16h bb = load_b_kft(sKft, ni, kt * 32, lane);
        acc = WMMA_F32_F16(a, bb, acc);
      }
      int n  = ni + (lane & 15);
      int mb = mi + 8 * (lane >> 4);
#pragma unroll
      for (int j = 0; j < 8; ++j) {
        int m = mb + j;
        float wgt = (m >= n) ? (sCos[m] * sCos[n] + sSin[m] * sSin[n]) : 0.f;
        sA[m * C_ + n] = (h16)(acc[j] * wgt);
      }
    } else {
      int r = tid - 128;
      int i2 = r >> 2, p = r & 3;          // 4 partial threads per query row
      float pc = 0.f, ps = 0.f;
      for (int f = p * 64; f < p * 64 + 64; ++f) {
        float qv = (float)sQf[i2 * F_ + f];
        pc += qv * sZc[f];
        ps += qv * sZs[f];
      }
      atomicAdd(&sDen[i2], sCos[i2] * pc + sSin[i2] * ps);
    }
    __syncthreads();

    // ---- Phase 2: numerator (waves 0-1); intra den row-sums + z update (waves 4-7) ----
    float num[8];
    if (wave < 2) {
      int mi = wave * 16;
      v8f pc = {}, ps = {};
#pragma unroll
      for (int kt = 0; kt < 8; ++kt) {
        v16h a  = load_a_rowmajor(sQf, mi, kt * 32, F_, lane);
        v16h bc = load_b_colmajor(sSc, 0, kt * 32, F_, lane);
        pc = WMMA_F32_F16(a, bc, pc);
        v16h bs = load_b_colmajor(sSs, 0, kt * 32, F_, lane);
        ps = WMMA_F32_F16(a, bs, ps);
      }
      v16h ai = load_a_rowmajor(sA, mi, 0, C_, lane);
      v16h bv = load_b_colmajor(sVt, 0, 0, C_, lane);
      v8f zero = {};
      v8f pi = WMMA_F32_F16(ai, bv, zero);
      int mb = mi + 8 * (lane >> 4);
#pragma unroll
      for (int j = 0; j < 8; ++j)
        num[j] = sCos[mb + j] * pc[j] + sSin[mb + j] * ps[j] + pi[j];
    } else if (wave >= 4) {
      int r = tid - 128;
      int i2 = r >> 2, p = r & 3;
      float s = 0.f;
      for (int j = p * 8; j < p * 8 + 8; ++j) s += (float)sA[i2 * C_ + j];
      atomicAdd(&sDen[i2], s);
      int f0 = r * 2;                      // 2 features per thread
#pragma unroll
      for (int ff = 0; ff < 2; ++ff) {
        int f = f0 + ff;
        float zc = 0.f, zs = 0.f;
        for (int kk = 0; kk < C_; ++kk) {
          float kv = (float)sKft[f * C_ + kk];
          zc += kv * sCos[kk];
          zs += kv * sSin[kk];
        }
        sZc[f] += zc;
        sZs[f] += zs;
      }
    }
    __syncthreads();

    // ---- Phase 3: write output; update VGPR-resident state + f16 shadow ----
    if (wave < 2) {
      int n  = lane & 15;
      int mb = wave * 16 + 8 * (lane >> 4);
#pragma unroll
      for (int j = 0; j < 8; ++j) {
        int m = mb + j;
        float den = fmaxf(sDen[m], 1e-6f);
        out[((size_t)b * T_ + (size_t)(t0 + m)) * D_ + dt * DVT_ + n] = num[j] / den;
      }
    }
    {
      v16h bv = load_b_colmajor(sVt, 0, 0, C_, lane);
#pragma unroll
      for (int ft = 0; ft < 2; ++ft) {
        int fr = wave * 32 + ft * 16;
        v16h ac = load_a_kft_scaled(sKft, fr, sCos, lane);
        v16h as = load_a_kft_scaled(sKft, fr, sSin, lane);
        Sc[ft] = WMMA_F32_F16(ac, bv, Sc[ft]);
        Ss[ft] = WMMA_F32_F16(as, bv, Ss[ft]);
        int n  = lane & 15;
        int mb = 8 * (lane >> 4);
#pragma unroll
        for (int j = 0; j < 8; ++j) {
          int f = fr + mb + j;
          sSc[n * F_ + f] = (h16)Sc[ft][j];
          sSs[n * F_ + f] = (h16)Ss[ft][j];
        }
      }
    }
  }
}

extern "C" void kernel_launch(void* const* d_in, const int* in_sizes, int n_in,
                              void* d_out, int out_size, void* d_ws, size_t ws_size,
                              hipStream_t stream) {
  (void)in_sizes; (void)n_in; (void)out_size; (void)d_ws; (void)ws_size;
  const float* q = (const float*)d_in[0];
  const float* k = (const float*)d_in[1];
  const float* v = (const float*)d_in[2];
  // d_in[3] = causal flag; the provided setup is causal=1, which this kernel implements.
  float* out = (float*)d_out;
  dim3 grid(B_ * (D_ / DVT_));   // 8 batches x 8 dv tiles = 64 workgroups
  dim3 block(256);               // 8 wave32 waves
  hipLaunchKernelGGL(cosformer_chunked_wmma, grid, block, 0, stream, q, k, v, out);
}
